// SwinBlock_7791070675427
// MI455X (gfx1250) — matrix operand order repaired and checked by reading, not compiled, not run on hardware
//
#include <hip/hip_runtime.h>
#include <hip/hip_bf16.h>
#include <stdint.h>

typedef __attribute__((ext_vector_type(8)))  _Float16 v8h;
typedef __attribute__((ext_vector_type(16))) _Float16 v16h;
typedef __attribute__((ext_vector_type(8)))  float    v8f;

#define DIM    192
#define HEADS  6
#define NTOK   49
#define BW     4096
#define HIDDEN 768
#define MROWS  (BW * NTOK)   // 200704

// ---------------------------------------------------------------------------
// A-fragment loader for V_WMMA_F32_16X16X32_F16.
// 16-bit A 16x32 layout (contiguous, matches B): lanes 0-15 hold K 0..15,
// lanes 16-31 hold K 16..31 for row M=lane%16.  One 32B chunk.
// ---------------------------------------------------------------------------
__device__ __forceinline__ v16h load_afrag(const _Float16* rowp, int lane) {
    return *(const v16h*)(rowp + (lane >> 4) * 16);
}

// B-fragment: B is 32(K)x16(N); lane l holds column n=l%16, K-range
// (l/16)*16..+15 -> contiguous 32B from B^T (stored as (N,K) row-major).
__device__ __forceinline__ v16h load_bfrag(const _Float16* colp, int lane) {
    return *(const v16h*)(colp + (lane >> 4) * 16);
}

__device__ __forceinline__ v8f wmma_f16(v16h a, v16h b, v8f c) {
    return __builtin_amdgcn_wmma_f32_16x16x32_f16(false, a, false, b,
                                                  (short)0, c, false, false);
}

// ---------------------------------------------------------------------------
// Prep: transpose weights to (N,K) f16; gather relative-position bias into a
// padded (HEADS,64,64) table with -1e30 in padded columns (softmax mask) and
// 0 in padded rows -> the attention kernel needs no guards at all.
// ---------------------------------------------------------------------------
__global__ __launch_bounds__(256) void prep_kernel(
    const float* __restrict__ qkv_w,  const float* __restrict__ proj_w,
    const float* __restrict__ fc1_w,  const float* __restrict__ fc2_w,
    const float* __restrict__ bias_table, const int* __restrict__ rel_index,
    _Float16* __restrict__ wqkvT, _Float16* __restrict__ wprojT,
    _Float16* __restrict__ wfc1T, _Float16* __restrict__ wfc2T,
    float* __restrict__ biasPad)
{
    int i = blockIdx.x * 256 + threadIdx.x;
    if (i < 576 * DIM) { int o = i / DIM, k = i % DIM; wqkvT[i] = (_Float16)qkv_w[k * 576 + o]; }
    if (i < DIM * DIM) { int o = i / DIM, k = i % DIM; wprojT[i] = (_Float16)proj_w[k * DIM + o]; }
    if (i < HIDDEN * DIM) { int o = i / DIM, k = i % DIM; wfc1T[i] = (_Float16)fc1_w[k * HIDDEN + o]; }
    if (i < DIM * HIDDEN) { int o = i / HIDDEN, k = i % HIDDEN; wfc2T[i] = (_Float16)fc2_w[k * DIM + o]; }
    if (i < HEADS * 64 * 64) {
        int h = i >> 12, rem = i & 4095;
        int row = rem >> 6, col = rem & 63;
        float v;
        if (col >= NTOK)      v = -1e30f;                 // masked column
        else if (row >= NTOK) v = 0.f;                    // padded row (discarded)
        else                  v = bias_table[rel_index[row * NTOK + col] * HEADS + h];
        biasPad[i] = v;
    }
}

// ---------------------------------------------------------------------------
// LayerNorm over DIM=192, f32 in -> f16 out.  One wave32 per row.
// ---------------------------------------------------------------------------
__global__ __launch_bounds__(256) void ln_kernel(
    const float* __restrict__ x, const float* __restrict__ g,
    const float* __restrict__ b, _Float16* __restrict__ out, int mrows)
{
    int wave = threadIdx.x >> 5, lane = threadIdx.x & 31;
    int row = blockIdx.x * 8 + wave;
    if (row >= mrows) return;
    const float* xr = x + (size_t)row * DIM;
    float v[6], s = 0.f, sq = 0.f;
#pragma unroll
    for (int i = 0; i < 6; ++i) { v[i] = xr[i * 32 + lane]; s += v[i]; sq += v[i] * v[i]; }
#pragma unroll
    for (int m = 16; m >= 1; m >>= 1) { s += __shfl_xor(s, m, 32); sq += __shfl_xor(sq, m, 32); }
    float mean = s * (1.f / DIM);
    float var  = sq * (1.f / DIM) - mean * mean;
    float inv  = rsqrtf(var + 1e-5f);
    _Float16* orow = out + (size_t)row * DIM;
#pragma unroll
    for (int i = 0; i < 6; ++i) {
        int c = i * 32 + lane;
        orow[c] = (_Float16)((v[i] - mean) * inv * g[c] + b[c]);
    }
}

// ---------------------------------------------------------------------------
// Generic f16 WMMA GEMM:  C(M,Nc) = epi( A(M,K) @ Bt(Nc,K)^T )
// EPI 0: store f16                       (QKV)
// EPI 1: +bias, exact GELU, store f16    (FC1)
// EPI 2: +bias +residual, store f32      (proj, FC2)
// Block = 256 threads (8 waves) -> 64x64 tile; A staged in LDS per K-step.
// ---------------------------------------------------------------------------
template <int EPI>
__global__ __launch_bounds__(256) void gemm_kernel(
    const _Float16* __restrict__ A, const _Float16* __restrict__ Bt,
    const float* __restrict__ bias, const float* __restrict__ res,
    void* __restrict__ Cout, int M, int K, int Nc)
{
    __shared__ alignas(128) _Float16 As[64][32];
    int tid = threadIdx.x, wave = tid >> 5, lane = tid & 31;
    int rowBlk = blockIdx.x * 64, colBlk = blockIdx.y * 64;
    int mt = wave >> 1;              // 0..3 row tile
    int ntBase = (wave & 1) * 2;     // 0 or 2 col tile pair
    v8f acc0 = {}, acc1 = {};

    int aRow  = mt * 16 + (lane & 15);
    int bRow0 = colBlk + ntBase * 16 + (lane & 15);
    int bRow1 = bRow0 + 16;
    int stRow = tid >> 2, stCol = (tid & 3) * 8;          // 64x32 staging, 16B/thread
    const _Float16* Arow = A + (size_t)(rowBlk + stRow) * K + stCol;
    const _Float16* Bp0  = Bt + (size_t)bRow0 * K;
    const _Float16* Bp1  = Bt + (size_t)bRow1 * K;

    for (int k0 = 0; k0 < K; k0 += 32) {
        __syncthreads();
        *(uint4*)&As[stRow][stCol] = *(const uint4*)(Arow + k0);
        __builtin_prefetch(Arow + k0 + 32, 0, 0);          // global_prefetch_b8
        __syncthreads();
        v16h af  = load_afrag(&As[aRow][0], lane);
        v16h bf0 = load_bfrag(Bp0 + k0, lane);
        v16h bf1 = load_bfrag(Bp1 + k0, lane);
        acc0 = wmma_f16(af, bf0, acc0);
        acc1 = wmma_f16(af, bf1, acc1);
    }

    int outRow = rowBlk + mt * 16 + (lane >> 4) * 8;
    int col0 = colBlk + ntBase * 16 + (lane & 15);
    int col1 = col0 + 16;
#pragma unroll
    for (int r = 0; r < 8; ++r) {
        float v0 = acc0[r], v1 = acc1[r];
        size_t row = (size_t)(outRow + r);
        if (EPI == 0) {
            _Float16* C = (_Float16*)Cout;
            C[row * Nc + col0] = (_Float16)v0;
            C[row * Nc + col1] = (_Float16)v1;
        } else if (EPI == 1) {
            v0 += bias[col0]; v1 += bias[col1];
            v0 = 0.5f * v0 * (1.0f + erff(v0 * 0.70710678118654752f));
            v1 = 0.5f * v1 * (1.0f + erff(v1 * 0.70710678118654752f));
            _Float16* C = (_Float16*)Cout;
            C[row * Nc + col0] = (_Float16)v0;
            C[row * Nc + col1] = (_Float16)v1;
        } else {
            float* C = (float*)Cout;
            size_t i0 = row * Nc + col0, i1 = row * Nc + col1;
            C[i0] = v0 + bias[col0] + res[i0];
            C[i1] = v1 + bias[col1] + res[i1];
        }
    }
}

// ---------------------------------------------------------------------------
// Windowed attention, one 128-thread block per (window b, head h).
// scores = (q*scale) @ k^T + biasPad -> softmax -> @ v.  49 padded to 64;
// the padded bias table already carries the -1e30 column mask, so the whole
// softmax path is branch-free.
// ---------------------------------------------------------------------------
__global__ __launch_bounds__(128) void attn_kernel(
    const _Float16* __restrict__ qkv,     // (MROWS, 576): [q|k|v] x head x 32
    const float* __restrict__ biasPad,    // (HEADS, 64, 64), pre-masked
    _Float16* __restrict__ ao)            // (MROWS, 192)
{
    __shared__ alignas(128) _Float16 qs[64][32];
    __shared__ alignas(128) _Float16 ks[64][32];
    __shared__ alignas(128) _Float16 vts[32][64];   // v transposed: [d][token]
    __shared__ alignas(128) _Float16 ps[64][64];    // softmax probs
    int b = blockIdx.x, h = blockIdx.y;
    int tid = threadIdx.x, wave = tid >> 5, lane = tid & 31;

    // ---- stage q, k, v (padded rows zeroed) ----
    {
        int r = tid >> 1, part = tid & 1, c = part * 16;
        v16h zero = {};
        v16h qv = zero, kv = zero, vv = zero;
        if (r < NTOK) {
            const _Float16* base = qkv + ((size_t)b * NTOK + r) * 576 + h * 32 + c;
            qv = *(const v16h*)(base);
            kv = *(const v16h*)(base + 192);
            vv = *(const v16h*)(base + 384);
        }
        *(v16h*)&qs[r][c] = qv;
        *(v16h*)&ks[r][c] = kv;
#pragma unroll
        for (int j = 0; j < 16; ++j) vts[c + j][r] = vv[j];
    }
    __syncthreads();

    // ---- scores: wave `wave` owns 16-row strip, all 64 cols ----
    int mt = wave;
    v16h qf = load_afrag(&qs[mt * 16 + (lane & 15)][0], lane);
    v8f sacc[4];
#pragma unroll
    for (int nt = 0; nt < 4; ++nt) {
        v16h kf = load_bfrag(&ks[nt * 16 + (lane & 15)][0], lane);
        v8f z = {};
        sacc[nt] = wmma_f16(qf, kf, z);
    }

    // ---- softmax (f32, branch-free, shfl across 16-lane column groups) ----
    const float scale = 0.17677669529663687f;   // 1/sqrt(32)
    int rBase = mt * 16 + (lane >> 4) * 8;
    int cLane = lane & 15;
    const float* bp = biasPad + ((size_t)h << 12) + (rBase << 6) + cLane;
    float s[4][8];
#pragma unroll
    for (int nt = 0; nt < 4; ++nt)
#pragma unroll
        for (int r = 0; r < 8; ++r)
            s[nt][r] = sacc[nt][r] * scale + bp[(r << 6) + nt * 16];

#pragma unroll
    for (int r = 0; r < 8; ++r) {
        float mx = fmaxf(fmaxf(s[0][r], s[1][r]), fmaxf(s[2][r], s[3][r]));
#pragma unroll
        for (int m = 8; m >= 1; m >>= 1) mx = fmaxf(mx, __shfl_xor(mx, m, 32));
        float e[4], sum = 0.f;
#pragma unroll
        for (int nt = 0; nt < 4; ++nt) { e[nt] = __expf(s[nt][r] - mx); sum += e[nt]; }
#pragma unroll
        for (int m = 8; m >= 1; m >>= 1) sum += __shfl_xor(sum, m, 32);
        float inv = 1.f / sum;
        int row = rBase + r;
#pragma unroll
        for (int nt = 0; nt < 4; ++nt)
            ps[row][nt * 16 + cLane] = (_Float16)(e[nt] * inv);
    }
    // each wave reads only its own strip of ps -> no cross-wave barrier needed

    // ---- out = P(16x64) @ V(64x32) : 2 K-steps x 2 col tiles ----
    v8f o0 = {}, o1 = {};
#pragma unroll
    for (int kt = 0; kt < 2; ++kt) {
        v16h pf  = load_afrag(&ps[mt * 16 + (lane & 15)][kt * 32], lane);
        v16h vf0 = load_bfrag(&vts[cLane][kt * 32], lane);
        v16h vf1 = load_bfrag(&vts[16 + cLane][kt * 32], lane);
        o0 = wmma_f16(pf, vf0, o0);
        o1 = wmma_f16(pf, vf1, o1);
    }
#pragma unroll
    for (int r = 0; r < 8; ++r) {
        int tok = rBase + r;
        if (tok < NTOK) {
            size_t base = ((size_t)b * NTOK + tok) * DIM + h * 32;
            ao[base + cLane]      = (_Float16)o0[r];
            ao[base + 16 + cLane] = (_Float16)o1[r];
        }
    }
}

// ---------------------------------------------------------------------------
extern "C" void kernel_launch(void* const* d_in, const int* in_sizes, int n_in,
                              void* d_out, int out_size, void* d_ws, size_t ws_size,
                              hipStream_t stream) {
    const float* x          = (const float*)d_in[0];
    const float* ln1_g      = (const float*)d_in[1];
    const float* ln1_b      = (const float*)d_in[2];
    const float* qkv_w      = (const float*)d_in[3];
    const float* proj_w     = (const float*)d_in[4];
    const float* proj_b     = (const float*)d_in[5];
    const float* bias_table = (const float*)d_in[6];
    const float* ln2_g      = (const float*)d_in[7];
    const float* ln2_b      = (const float*)d_in[8];
    const float* fc1_w      = (const float*)d_in[9];
    const float* fc1_b      = (const float*)d_in[10];
    const float* fc2_w      = (const float*)d_in[11];
    const float* fc2_b      = (const float*)d_in[12];
    const int*   rel_index  = (const int*)d_in[13];

    char* w = (char*)d_ws;
    auto carve = [&](size_t bytes) -> char* {
        char* p = w; w += (bytes + 255) & ~(size_t)255; return p;
    };
    _Float16* wqkvT   = (_Float16*)carve((size_t)576 * DIM * 2);
    _Float16* wprojT  = (_Float16*)carve((size_t)DIM * DIM * 2);
    _Float16* wfc1T   = (_Float16*)carve((size_t)HIDDEN * DIM * 2);
    _Float16* wfc2T   = (_Float16*)carve((size_t)DIM * HIDDEN * 2);
    float*    biasPad = (float*)   carve((size_t)HEADS * 64 * 64 * 4);
    _Float16* xh      = (_Float16*)carve((size_t)MROWS * DIM * 2);     // LN out (reused)
    _Float16* big     = (_Float16*)carve((size_t)MROWS * HIDDEN * 2);  // qkv / fc1 out
    _Float16* ao      = (_Float16*)carve((size_t)MROWS * DIM * 2);     // attention out
    float*    x1      = (float*)   carve((size_t)MROWS * DIM * 4);     // residual 1
    (void)ws_size; (void)n_in; (void)in_sizes; (void)out_size;

    // 0) weight transpose/convert + padded bias gather
    prep_kernel<<<(HIDDEN * DIM + 255) / 256, 256, 0, stream>>>(
        qkv_w, proj_w, fc1_w, fc2_w, bias_table, rel_index,
        wqkvT, wprojT, wfc1T, wfc2T, biasPad);

    // 1) xh = LN1(x)
    ln_kernel<<<MROWS / 8, 256, 0, stream>>>(x, ln1_g, ln1_b, xh, MROWS);

    // 2) qkv = xh @ Wqkv   (f16 out, no bias)
    gemm_kernel<0><<<dim3(MROWS / 64, 576 / 64), 256, 0, stream>>>(
        xh, wqkvT, nullptr, nullptr, big, MROWS, DIM, 576);

    // 3) attention per (window, head)
    attn_kernel<<<dim3(BW, HEADS), 128, 0, stream>>>(big, biasPad, ao);

    // 4) x1 = x + ao @ Wproj + b
    gemm_kernel<2><<<dim3(MROWS / 64, DIM / 64), 256, 0, stream>>>(
        ao, wprojT, proj_b, x, x1, MROWS, DIM, DIM);

    // 5) xh = LN2(x1)
    ln_kernel<<<MROWS / 8, 256, 0, stream>>>(x1, ln2_g, ln2_b, xh, MROWS);

    // 6) big = gelu(xh @ Wfc1 + b)
    gemm_kernel<1><<<dim3(MROWS / 64, HIDDEN / 64), 256, 0, stream>>>(
        xh, wfc1T, fc1_b, nullptr, big, MROWS, DIM, HIDDEN);

    // 7) out = x1 + big @ Wfc2 + b
    gemm_kernel<2><<<dim3(MROWS / 64, DIM / 64), 256, 0, stream>>>(
        big, wfc2T, fc2_b, x1, (float*)d_out, MROWS, HIDDEN, DIM);
}
